// DecoderBlock_90288802497026
// MI455X (gfx1250) — matrix operand order repaired and checked by reading, not compiled, not run on hardware
//
#include <hip/hip_runtime.h>
#include <cstdint>
#include <cstddef>

// ---------------------------------------------------------------------------
// CDNA5 (gfx1250) decoder block: bf16 WMMA GEMMs + WMMA flash attention.
// wave32 everywhere; WMMA layouts per CDNA5 ISA 7.12.2.
// Global->LDS staging uses GLOBAL_LOAD_ASYNC_TO_LDS_B128 (ASYNCcnt) when the
// toolchain exposes the builtin; falls back to plain LDS stores otherwise.
// ---------------------------------------------------------------------------

typedef __bf16 bf16;
typedef __attribute__((ext_vector_type(16))) __bf16 v16bf;
typedef __attribute__((ext_vector_type(8)))  float  v8f;

#define DEVINL static __device__ __forceinline__

#if defined(__gfx1250__) && __has_builtin(__builtin_amdgcn_global_load_async_to_lds_b128)
#define ASYNC_LDS 1
#else
#define ASYNC_LDS 0
#endif

DEVINL bf16 f2bf(float f) {
  uint32_t u = __builtin_bit_cast(uint32_t, f);
  u += 0x7FFFu + ((u >> 16) & 1u);          // round-to-nearest-even
  uint16_t s = (uint16_t)(u >> 16);
  return __builtin_bit_cast(bf16, s);
}

struct uint4x2 { uint4 lo, hi; };
DEVINL v16bf ld_frag(const bf16* lo, const bf16* hi) {
  uint4x2 p;
  p.lo = *reinterpret_cast<const uint4*>(lo);
  p.hi = *reinterpret_cast<const uint4*>(hi);
  return __builtin_bit_cast(v16bf, p);
}

DEVINL v8f wmma_bf16(v16bf a, v16bf b, v8f c) {
  // D = A(16x32) * B(32x16) + C(16x16), f32 accumulate
  return __builtin_amdgcn_wmma_f32_16x16x32_bf16(false, a, false, b, (short)0, c,
                                                 false, false);
}

#if ASYNC_LDS
// exact builtin parameter types: (AS1 v4i*, AS3 v4i*, imm int, imm int)
typedef int v4i_async __attribute__((vector_size(16)));
typedef __attribute__((address_space(1))) v4i_async* as1_v4i_p;
typedef __attribute__((address_space(3))) v4i_async* as3_v4i_p;
#endif

// 16-byte global -> LDS copy (async on gfx1250, sync fallback elsewhere)
DEVINL void cp16(bf16* l, const bf16* g) {
#if ASYNC_LDS
  __builtin_amdgcn_global_load_async_to_lds_b128((as1_v4i_p)g, (as3_v4i_p)l, 0, 0);
#else
  *(uint4*)l = *(const uint4*)g;
#endif
}

DEVINL void cp_wait() {
#if ASYNC_LDS
#if __has_builtin(__builtin_amdgcn_s_wait_asynccnt)
  __builtin_amdgcn_s_wait_asynccnt(0);
#else
  asm volatile("s_wait_asynccnt 0x0" ::: "memory");
#endif
#endif
}

DEVINL float gelu_exact(float x) {
  return 0.5f * x * (1.0f + erff(x * 0.7071067811865476f));
}

// ---------------------------------------------------------------------------
// Weight convert + transpose:  W[din][dout] f32  ->  Wt[dout][din] bf16
// ---------------------------------------------------------------------------
__global__ __launch_bounds__(256) void wtrans_kernel(const float* __restrict__ W,
                                                     bf16* __restrict__ Wt,
                                                     int din, int dout) {
  size_t idx = (size_t)blockIdx.x * 256 + threadIdx.x;
  size_t n = (size_t)din * dout;
  if (idx >= n) return;
  int o = (int)(idx / din);
  int i = (int)(idx % din);
  Wt[idx] = f2bf(W[(size_t)i * dout + o]);
}

// f32 -> bf16 flat convert
__global__ __launch_bounds__(256) void cvt_kernel(const float* __restrict__ in,
                                                  bf16* __restrict__ out, int n) {
  int idx = blockIdx.x * 256 + threadIdx.x;
  if (idx < n) out[idx] = f2bf(in[idx]);
}

// ---------------------------------------------------------------------------
// LayerNorm over 1024 cols, fp32 in -> bf16 out.  grid = rows, block = 128.
// ---------------------------------------------------------------------------
__global__ __launch_bounds__(128) void ln_kernel(const float* __restrict__ X,
                                                 const float* __restrict__ w,
                                                 const float* __restrict__ bparm,
                                                 bf16* __restrict__ Y) {
  const int row = blockIdx.x;
  const int tid = threadIdx.x;
  const int lane = tid & 31, wave = tid >> 5;
  const float* xr = X + (size_t)row * 1024;
  float vloc[8], s = 0.f, q = 0.f;
#pragma unroll
  for (int i = 0; i < 8; ++i) {
    float v = xr[tid + 128 * i];
    vloc[i] = v; s += v; q += v * v;
  }
#pragma unroll
  for (int off = 16; off >= 1; off >>= 1) {
    s += __shfl_xor(s, off, 32);
    q += __shfl_xor(q, off, 32);
  }
  __shared__ float ss[4], qq[4];
  if (lane == 0) { ss[wave] = s; qq[wave] = q; }
  __syncthreads();
  s = ss[0] + ss[1] + ss[2] + ss[3];
  q = qq[0] + qq[1] + qq[2] + qq[3];
  const float mean = s * (1.f / 1024.f);
  const float var = q * (1.f / 1024.f) - mean * mean;
  const float rs = rsqrtf(var + 1e-5f);
  bf16* yr = Y + (size_t)row * 1024;
#pragma unroll
  for (int i = 0; i < 8; ++i) {
    int c = tid + 128 * i;
    yr[c] = f2bf((vloc[i] - mean) * rs * w[c] + bparm[c]);
  }
}

// ---------------------------------------------------------------------------
// Generic bf16 WMMA GEMM.  A[M][K] bf16 row-major, Bt[N][K] bf16 row-major
// (pre-transposed weights).  C = A*B + bias (+resid) (GELU), f32/bf16 out.
// Block 256 thr = 8 waves, tile 128x128, K-step 32, double-buffered LDS with
// async global->LDS staging overlapped with WMMA compute.
// Requires M%128==0, N%128==0, K%32==0 (holds for all call sites).
// ---------------------------------------------------------------------------
template <bool RESID, bool GELU_, bool WF32, bool WBF16>
__global__ __launch_bounds__(256) void gemm_kernel(const bf16* __restrict__ A,
                                                   const bf16* __restrict__ Bt,
                                                   const float* __restrict__ bias,
                                                   const float* __restrict__ resid,
                                                   float* __restrict__ Cf,
                                                   bf16* __restrict__ Cb,
                                                   int N, int K) {
  __shared__ bf16 a_tile[2][128 * 40];  // 32-half rows, +8 pad
  __shared__ bf16 b_tile[2][128 * 40];  // column-major tile of B (rows of Bt)

  const int tid = threadIdx.x;
  const int lane = tid & 31, ln = lane & 15, hh = lane >> 4;
  const int wave = tid >> 5, wm = wave >> 2, wn = wave & 3;
  const size_t rowBase = (size_t)blockIdx.y * 128;
  const size_t colBase = (size_t)blockIdx.x * 128;

  v8f acc[4][2] = {};

  auto stage = [&](int bufi, int k0) {
#pragma unroll
    for (int c = tid; c < 512; c += 256) {
      int r = c >> 2, p = c & 3;
      cp16(a_tile[bufi] + r * 40 + p * 8,
           A + (rowBase + r) * (size_t)K + k0 + p * 8);
      cp16(b_tile[bufi] + r * 40 + p * 8,
           Bt + (colBase + r) * (size_t)K + k0 + p * 8);
    }
  };

  int buf = 0;
  stage(0, 0);
  cp_wait();
  __syncthreads();

  for (int k0 = 0; k0 < K; k0 += 32) {
    if (k0 + 32 < K) stage(buf ^ 1, k0 + 32);  // overlap next-tile fill

    v16bf af[4], bfr[2];
#pragma unroll
    for (int i = 0; i < 4; ++i) {
      const bf16* ap = a_tile[buf] + (wm * 64 + 16 * i + ln) * 40 + 8 * hh;
      af[i] = ld_frag(ap, ap + 16);  // K 8h..8h+7 | 16+8h..16+8h+7
    }
#pragma unroll
    for (int j = 0; j < 2; ++j) {
      const bf16* bp = b_tile[buf] + (wn * 32 + 16 * j + ln) * 40 + 8 * hh;
      bfr[j] = ld_frag(bp, bp + 16);  // K 8h..8h+7 | 16+8h..16+8h+7
    }
#pragma unroll
    for (int i = 0; i < 4; ++i)
#pragma unroll
      for (int j = 0; j < 2; ++j) acc[i][j] = wmma_bf16(af[i], bfr[j], acc[i][j]);

    cp_wait();        // next tile landed in LDS
    __syncthreads();  // all waves done with current buffer
    buf ^= 1;
  }

  // epilogue
#pragma unroll
  for (int i = 0; i < 4; ++i) {
#pragma unroll
    for (int j = 0; j < 2; ++j) {
      size_t col = colBase + wn * 32 + 16 * j + ln;
      float bcol = bias[col];
#pragma unroll
      for (int e = 0; e < 8; ++e) {
        size_t row = rowBase + wm * 64 + 16 * i + e + 8 * hh;
        float v = acc[i][j][e] + bcol;
        if (RESID) v += resid[row * N + col];
        if (GELU_) v = gelu_exact(v);
        if (WF32) Cf[row * N + col] = v;
        if (WBF16) Cb[row * N + col] = f2bf(v);
      }
    }
  }
}

// ---------------------------------------------------------------------------
// qkv f32 [B,T,3,H,64] -> Q/K bf16 [B,H,T,64], Vt bf16 [B,H,64,T]
// ---------------------------------------------------------------------------
__global__ __launch_bounds__(256) void split_qkv_kernel(const float* __restrict__ qkv,
                                                        bf16* __restrict__ Qb,
                                                        bf16* __restrict__ Kb,
                                                        bf16* __restrict__ Vt) {
  int idx = blockIdx.x * 256 + threadIdx.x;  // B*T*H*64 = 2^22
  int d = idx & 63, h = (idx >> 6) & 15, t = (idx >> 10) & 1023, b = idx >> 20;
  size_t base = ((size_t)(b * 1024 + t)) * 3072;
  float qv = qkv[base + h * 64 + d];
  float kv = qkv[base + 1024 + h * 64 + d];
  float vv = qkv[base + 2048 + h * 64 + d];
  size_t bh = (size_t)(b * 16 + h);
  Qb[(bh * 1024 + t) * 64 + d] = f2bf(qv);
  Kb[(bh * 1024 + t) * 64 + d] = f2bf(kv);
  Vt[(bh * 64 + d) * 1024 + t] = f2bf(vv);
}

// q f32 [B,T,H*64] -> Qb bf16 [B,H,T,64]
__global__ __launch_bounds__(256) void repack_q_kernel(const float* __restrict__ in,
                                                       bf16* __restrict__ Qb) {
  int idx = blockIdx.x * 256 + threadIdx.x;  // 2^22
  int c = idx & 1023, t = (idx >> 10) & 1023, b = idx >> 20;
  int h = c >> 6, d = c & 63;
  Qb[(((size_t)(b * 16 + h)) * 1024 + t) * 64 + d] = f2bf(in[idx]);
}

// k/v f32 [B,S,H*64] -> Kb bf16 [B,H,S,64], Vt bf16 [B,H,64,S]
__global__ __launch_bounds__(256) void repack_kv_kernel(const float* __restrict__ kin,
                                                        const float* __restrict__ vin,
                                                        bf16* __restrict__ Kb,
                                                        bf16* __restrict__ Vt, int S) {
  int idx = blockIdx.x * 256 + threadIdx.x;
  int n = 4 * S * 1024;
  if (idx >= n) return;
  int c = idx & 1023;
  int s = (idx >> 10) % S;
  int b = (idx >> 10) / S;
  int h = c >> 6, d = c & 63;
  size_t bh = (size_t)(b * 16 + h);
  Kb[(bh * S + s) * 64 + d] = f2bf(kin[idx]);
  Vt[(bh * 64 + d) * (size_t)S + s] = f2bf(vin[idx]);
}

// ---------------------------------------------------------------------------
// Flash attention (WMMA, online softmax).  HD=64.
// grid = (T/64, B*H), block = 128 (4 waves, 16 query rows each).
// Q [B,H,T,64], K [B,H,S,64], Vt [B,H,64,S], out bf16 [B*T, H*64].
// ---------------------------------------------------------------------------
template <bool CAUSAL>
__global__ __launch_bounds__(128) void flash_kernel(const bf16* __restrict__ Q,
                                                    const bf16* __restrict__ Kh,
                                                    const bf16* __restrict__ Vt,
                                                    const unsigned char* __restrict__ kmask,
                                                    bf16* __restrict__ Out,
                                                    int T, int Sk, int Hn) {
  __shared__ bf16 kbuf[32 * 64];    // [s_rel][d]
  __shared__ bf16 vbuf[64 * 32];    // [d][s_rel]
  __shared__ bf16 ptile[4 * 16 * 32];

  const int tid = threadIdx.x;
  const int lane = tid & 31, ln = lane & 15, hh = lane >> 4, wave = tid >> 5;
  const int tblk = blockIdx.x, bh = blockIdx.y;
  const int b = bh / Hn, h = bh % Hn;

  const bf16* Qb = Q + ((size_t)bh * T) * 64;
  const bf16* Kb = Kh + ((size_t)bh * Sk) * 64;
  const bf16* Vb = Vt + ((size_t)bh * 64) * Sk;
  const int tw = tblk * 64 + wave * 16;

  // Q fragments for this wave's 16 rows (A operand layout, d split 0-31/32-63)
  const bf16* qp = Qb + (size_t)(tw + ln) * 64;
  v16bf qf0 = ld_frag(qp + 8 * hh, qp + 16 + 8 * hh);
  v16bf qf1 = ld_frag(qp + 32 + 8 * hh, qp + 48 + 8 * hh);

  v8f o[4] = {};
  float mrow[8], lrow[8];
#pragma unroll
  for (int e = 0; e < 8; ++e) { mrow[e] = -3.0e38f; lrow[e] = 0.f; }
  const float scale = 0.125f;  // 64^-0.5
  const int Send = CAUSAL ? ((tblk * 64 + 64 < Sk) ? tblk * 64 + 64 : Sk) : Sk;
  bf16* pw = ptile + wave * 512;

  for (int s0 = 0; s0 < Send; s0 += 32) {
    __syncthreads();  // prior P*V reads of vbuf done before restage
#pragma unroll
    for (int c = tid; c < 256; c += 128) {  // K chunk: 32 rows x 64 d
      int s = c >> 3, p = c & 7;
      cp16(kbuf + s * 64 + p * 8, Kb + (size_t)(s0 + s) * 64 + p * 8);
    }
#pragma unroll
    for (int c = tid; c < 256; c += 128) {  // V chunk: 64 d x 32 s
      int d = c >> 2, p = c & 3;
      cp16(vbuf + d * 32 + p * 8, Vb + (size_t)d * Sk + s0 + p * 8);
    }
    cp_wait();
    __syncthreads();

    // S_tile = Q(16x64) * K^T(64x32): 2 N-tiles, K-dim split 32+32
    v8f sf[2] = {};
#pragma unroll
    for (int j = 0; j < 2; ++j) {
      const bf16* kp = kbuf + (16 * j + ln) * 64 + 8 * hh;
      v16bf b0 = ld_frag(kp, kp + 16);
      v16bf b1 = ld_frag(kp + 32, kp + 48);
      sf[j] = wmma_bf16(qf0, b0, sf[j]);
      sf[j] = wmma_bf16(qf1, b1, sf[j]);
    }

    // online softmax update (C layout: VGPR e -> row e+8*hh, lane%16 -> col)
#pragma unroll
    for (int e = 0; e < 8; ++e) {
      int trow = tw + e + 8 * hh;
      float v0 = sf[0][e] * scale;
      float v1 = sf[1][e] * scale;
      int c0 = s0 + ln, c1 = s0 + 16 + ln;
      bool m0 = (CAUSAL && c0 > trow) || kmask[b * Sk + c0];
      bool m1 = (CAUSAL && c1 > trow) || kmask[b * Sk + c1];
      if (m0) v0 = -3.0e38f;
      if (m1) v1 = -3.0e38f;
      float vm = fmaxf(v0, v1);
#pragma unroll
      for (int off = 8; off >= 1; off >>= 1) vm = fmaxf(vm, __shfl_xor(vm, off, 32));
      float mn = fmaxf(fmaxf(mrow[e], vm), -1.0e30f);
      float p0 = __expf(v0 - mn);
      float p1 = __expf(v1 - mn);
      float ps = p0 + p1;
#pragma unroll
      for (int off = 8; off >= 1; off >>= 1) ps += __shfl_xor(ps, off, 32);
      float f = __expf(mrow[e] - mn);
      lrow[e] = lrow[e] * f + ps;
      mrow[e] = mn;
#pragma unroll
      for (int dj = 0; dj < 4; ++dj) o[dj][e] *= f;
      pw[(e + 8 * hh) * 32 + ln] = f2bf(p0);
      pw[(e + 8 * hh) * 32 + 16 + ln] = f2bf(p1);
    }
    __syncthreads();  // ptile visible within wave/block

    // O += P(16x32) * V(32x64)
    const bf16* pa = pw + ln * 32 + 8 * hh;
    v16bf paf = ld_frag(pa, pa + 16);
#pragma unroll
    for (int dj = 0; dj < 4; ++dj) {
      const bf16* vp = vbuf + (16 * dj + ln) * 32 + 8 * hh;
      v16bf vbf = ld_frag(vp, vp + 16);
      o[dj] = wmma_bf16(paf, vbf, o[dj]);
    }
  }

  // normalize + store bf16 into [B*T, H*64]
  const size_t HD64 = (size_t)(Hn * 64);
#pragma unroll
  for (int e = 0; e < 8; ++e) {
    int trow = tw + e + 8 * hh;
    float rl = lrow[e] > 0.f ? 1.f / lrow[e] : 0.f;
#pragma unroll
    for (int dj = 0; dj < 4; ++dj)
      Out[((size_t)b * T + trow) * HD64 + h * 64 + 16 * dj + ln] =
          f2bf(o[dj][e] * rl);
  }
}

// ---------------------------------------------------------------------------
// Host orchestration
// ---------------------------------------------------------------------------
extern "C" void kernel_launch(void* const* d_in, const int* in_sizes, int n_in,
                              void* d_out, int out_size, void* d_ws, size_t ws_size,
                              hipStream_t stream) {
  (void)in_sizes; (void)n_in; (void)out_size; (void)ws_size;
  constexpr int B = 4, T = 1024, S = 576, DD = 1024, DE = 768, Hn = 16, DM = 4096;
  constexpr int MT = B * T;   // 4096
  constexpr int MS = B * S;   // 2304

  const float* x      = (const float*)d_in[0];
  const float* enc    = (const float*)d_in[1];
  const unsigned char* tmask = (const unsigned char*)d_in[2];
  const unsigned char* emask = (const unsigned char*)d_in[3];
  const float* ln1_w = (const float*)d_in[4],  *ln1_b = (const float*)d_in[5];
  const float* qkv_w = (const float*)d_in[6],  *qkv_b = (const float*)d_in[7];
  const float* proj_w = (const float*)d_in[8], *proj_b = (const float*)d_in[9];
  const float* ln2_w = (const float*)d_in[10], *ln2_b = (const float*)d_in[11];
  const float* q_w = (const float*)d_in[12],   *q_b = (const float*)d_in[13];
  const float* k_w = (const float*)d_in[14],   *k_b = (const float*)d_in[15];
  const float* v_w = (const float*)d_in[16],   *v_b = (const float*)d_in[17];
  const float* out_w = (const float*)d_in[18], *out_b = (const float*)d_in[19];
  const float* ln3_w = (const float*)d_in[20], *ln3_b = (const float*)d_in[21];
  const float* m1_w = (const float*)d_in[22],  *m1_b = (const float*)d_in[23];
  const float* m2_w = (const float*)d_in[24],  *m2_b = (const float*)d_in[25];
  float* outp = (float*)d_out;

  char* ws = (char*)d_ws;
  size_t off = 0;
  auto take = [&](size_t bytes) {
    size_t o = off;
    off += (bytes + 255) & ~(size_t)255;
    return o;
  };
  // persistent bf16 transposed weights
  size_t o_wqkv = take((size_t)DD * 3 * DD * 2);
  size_t o_wproj = take((size_t)DD * DD * 2);
  size_t o_wq = take((size_t)DD * DD * 2);
  size_t o_wk = take((size_t)DD * DE * 2);
  size_t o_wv = take((size_t)DD * DE * 2);
  size_t o_wout = take((size_t)DD * DD * 2);
  size_t o_wm1 = take((size_t)DM * DD * 2);
  size_t o_wm2 = take((size_t)DD * DM * 2);
  size_t o_encbf = take((size_t)MS * DE * 2);
  size_t o_x1 = take((size_t)MT * DD * 4);
  size_t o_x2 = take((size_t)MT * DD * 4);
  size_t o_arenaA = take((size_t)MT * 3 * DD * 4);          // 50.3 MB
  size_t o_arenaB = take(3 * (size_t)MT * DD * 2);          // 25.2 MB
  size_t o_arenaC = take((size_t)MT * DD * 2);              // 8.4 MB

  bf16* wt_qkv = (bf16*)(ws + o_wqkv);
  bf16* wt_proj = (bf16*)(ws + o_wproj);
  bf16* wt_q = (bf16*)(ws + o_wq);
  bf16* wt_k = (bf16*)(ws + o_wk);
  bf16* wt_v = (bf16*)(ws + o_wv);
  bf16* wt_out = (bf16*)(ws + o_wout);
  bf16* wt_m1 = (bf16*)(ws + o_wm1);
  bf16* wt_m2 = (bf16*)(ws + o_wm2);
  bf16* enc_bf = (bf16*)(ws + o_encbf);
  float* x1 = (float*)(ws + o_x1);
  float* x2 = (float*)(ws + o_x2);

  // arena A overlays
  float* qkvf = (float*)(ws + o_arenaA);                               // [4096,3072]
  float* qf32 = (float*)(ws + o_arenaA);                               // [4096,1024]
  float* kf32 = (float*)(ws + o_arenaA + (size_t)MT * DD * 4);         // [2304,1024]
  float* vf32 = (float*)(ws + o_arenaA + (size_t)MT * DD * 4 + (size_t)MS * DD * 4);
  bf16* gbf = (bf16*)(ws + o_arenaA);                                  // [4096,4096]
  // arena B overlays
  bf16* Qs = (bf16*)(ws + o_arenaB);
  bf16* Ks = (bf16*)(ws + o_arenaB + (size_t)MT * DD * 2);
  bf16* Vts = (bf16*)(ws + o_arenaB + 2 * (size_t)MT * DD * 2);
  bf16* Q2 = (bf16*)(ws + o_arenaB);
  bf16* K2 = (bf16*)(ws + o_arenaB + (size_t)MT * DD * 2);
  bf16* Vt2 = (bf16*)(ws + o_arenaB + (size_t)MT * DD * 2 + (size_t)MS * DD * 2);
  bf16* hbf3 = (bf16*)(ws + o_arenaB);
  // arena C (sequential reuse): ln1 out / sa / ln2 out / ca
  bf16* hbfC = (bf16*)(ws + o_arenaC);

  auto wgrid = [](size_t n) { return (unsigned)((n + 255) / 256); };

  // 1) weights -> bf16 transposed
  wtrans_kernel<<<wgrid((size_t)DD * 3 * DD), 256, 0, stream>>>(qkv_w, wt_qkv, DD, 3 * DD);
  wtrans_kernel<<<wgrid((size_t)DD * DD), 256, 0, stream>>>(proj_w, wt_proj, DD, DD);
  wtrans_kernel<<<wgrid((size_t)DD * DD), 256, 0, stream>>>(q_w, wt_q, DD, DD);
  wtrans_kernel<<<wgrid((size_t)DE * DD), 256, 0, stream>>>(k_w, wt_k, DE, DD);
  wtrans_kernel<<<wgrid((size_t)DE * DD), 256, 0, stream>>>(v_w, wt_v, DE, DD);
  wtrans_kernel<<<wgrid((size_t)DD * DD), 256, 0, stream>>>(out_w, wt_out, DD, DD);
  wtrans_kernel<<<wgrid((size_t)DD * DM), 256, 0, stream>>>(m1_w, wt_m1, DD, DM);
  wtrans_kernel<<<wgrid((size_t)DM * DD), 256, 0, stream>>>(m2_w, wt_m2, DM, DD);
  cvt_kernel<<<wgrid((size_t)MS * DE), 256, 0, stream>>>(enc, enc_bf, MS * DE);

  // 2) self-attention
  ln_kernel<<<MT, 128, 0, stream>>>(x, ln1_w, ln1_b, hbfC);
  gemm_kernel<false, false, true, false><<<dim3(3 * DD / 128, MT / 128), 256, 0, stream>>>(
      hbfC, wt_qkv, qkv_b, nullptr, qkvf, nullptr, 3 * DD, DD);
  split_qkv_kernel<<<wgrid((size_t)MT * DD), 256, 0, stream>>>(qkvf, Qs, Ks, Vts);
  flash_kernel<true><<<dim3(T / 64, B * Hn), 128, 0, stream>>>(Qs, Ks, Vts, tmask, hbfC,
                                                               T, T, Hn);
  gemm_kernel<true, false, true, false><<<dim3(DD / 128, MT / 128), 256, 0, stream>>>(
      hbfC, wt_proj, proj_b, x, x1, nullptr, DD, DD);

  // 3) cross-attention
  ln_kernel<<<MT, 128, 0, stream>>>(x1, ln2_w, ln2_b, hbfC);
  gemm_kernel<false, false, true, false><<<dim3(DD / 128, MT / 128), 256, 0, stream>>>(
      hbfC, wt_q, q_b, nullptr, qf32, nullptr, DD, DD);
  gemm_kernel<false, false, true, false><<<dim3(DD / 128, MS / 128), 256, 0, stream>>>(
      enc_bf, wt_k, k_b, nullptr, kf32, nullptr, DD, DE);
  gemm_kernel<false, false, true, false><<<dim3(DD / 128, MS / 128), 256, 0, stream>>>(
      enc_bf, wt_v, v_b, nullptr, vf32, nullptr, DD, DE);
  repack_q_kernel<<<wgrid((size_t)MT * DD), 256, 0, stream>>>(qf32, Q2);
  repack_kv_kernel<<<wgrid((size_t)MS * DD), 256, 0, stream>>>(kf32, vf32, K2, Vt2, S);
  flash_kernel<false><<<dim3(T / 64, B * Hn), 128, 0, stream>>>(Q2, K2, Vt2, emask, hbfC,
                                                                T, S, Hn);
  gemm_kernel<true, false, true, false><<<dim3(DD / 128, MT / 128), 256, 0, stream>>>(
      hbfC, wt_out, out_b, x1, x2, nullptr, DD, DD);

  // 4) MLP
  ln_kernel<<<MT, 128, 0, stream>>>(x2, ln3_w, ln3_b, hbf3);
  gemm_kernel<false, true, false, true><<<dim3(DM / 128, MT / 128), 256, 0, stream>>>(
      hbf3, wt_m1, m1_b, nullptr, nullptr, gbf, DM, DD);
  gemm_kernel<true, false, true, false><<<dim3(DD / 128, MT / 128), 256, 0, stream>>>(
      gbf, wt_m2, m2_b, x2, outp, nullptr, DD, DM);
}
